// Model_3478923510595
// MI455X (gfx1250) — compile-verified, matrix operand order checked by reading, not run on hardware
//
#include <hip/hip_runtime.h>
#include <hip/hip_bf16.h>

// ---------------------------------------------------------------------------
// MeshGraphNets step for MI455X (gfx1250, wave32, WMMA 16x16x32 f16).
// All GEMMs (encoders, 15x edge/node blocks, decoder) run through one fused
// 3-layer MLP kernel using v_wmma_f32_16x16x32_f16 with fp32 accumulation.
// Operand staging uses GLOBAL_LOAD_ASYNC_TO_LDS (ASYNCcnt). All 8 B-fragments
// are loaded into distinct registers before the WMMA burst so the 8 WMMAs
// issue back-to-back behind a single dscnt drain.
// ---------------------------------------------------------------------------

typedef _Float16 half8  __attribute__((ext_vector_type(8)));
typedef _Float16 half16 __attribute__((ext_vector_type(16)));
typedef float    floatx8 __attribute__((ext_vector_type(8)));
typedef int      int4v   __attribute__((ext_vector_type(4)));

#define HID   128
#define APAD  40    // chunk row stride (halves): 20 dwords -> conflict-free
#define HPAD  136   // hidden row stride (halves): 68 dwords -> conflict-free
#define LNEPS 1e-5f

#if defined(__has_builtin)
#if __has_builtin(__builtin_amdgcn_global_load_async_to_lds_b128) && \
    __has_builtin(__builtin_amdgcn_s_wait_asynccnt)
#define HAVE_ASYNC_LDS 1
#endif
#endif
#ifndef HAVE_ASYNC_LDS
#define HAVE_ASYNC_LDS 0
#endif

__device__ __forceinline__ void copy16ToLds(const void* g, void* l) {
#if HAVE_ASYNC_LDS
  // Signature (from compiler diagnostic): (AS1 int4*, AS3 int4*, imm, imm)
  __builtin_amdgcn_global_load_async_to_lds_b128(
      (__attribute__((address_space(1))) int4v*)(g),
      (__attribute__((address_space(3))) int4v*)(l), 0, 0);
#else
  *(half8*)l = *(const half8*)g;
#endif
}

__device__ __forceinline__ void stageWait() {
#if HAVE_ASYNC_LDS
  __builtin_amdgcn_s_wait_asynccnt(0);
#endif
}

struct MlpArgs {
  const _Float16* seg0;   // rows [k<128], identity indexed, stride 128
  const _Float16* seg1;   // rows [128<=k<256]
  const _Float16* seg2;   // rows [256<=k<384]
  const int* idx1;        // gather index for seg1 (null -> identity)
  const int* idx2;        // gather index for seg2 (null -> identity)
  const _Float16 *W1t, *W2t, *W3t;   // transposed f16 weights [128][KIN]/[128][128]
  const float *b1, *b2, *b3;
  const float *g, *beta;  // null -> no LayerNorm
  const float* resid;     // null -> no residual
  float* outF;            // [M][128] fp32 output
  _Float16* outH;         // [M][128] f16 copy (may be null)
  int KIN;                // 32 / 128 / 256 / 384
  int M;
};

__device__ __forceinline__ half16 ldsFrag(const _Float16* p) {
  union { half16 v; half8 h[2]; } u;
  u.h[0] = *(const half8*)(p);
  u.h[1] = *(const half8*)(p + 16);
  return u.v;
}

__global__ __launch_bounds__(256) void mlp3_wmma(MlpArgs a) {
  __shared__ __align__(16) _Float16 sA[128 * APAD];
  __shared__ __align__(16) _Float16 sW[128 * APAD];
  __shared__ __align__(16) _Float16 sH[128 * HPAD];
  __shared__ float sB[128], sG[128], sBt[128];

  const int tid  = threadIdx.x;
  const int lane = tid & 31;
  const int wave = tid >> 5;
  const int lr   = lane & 15;           // row-in-strip (A) / col-in-tile (B)
  const int kh   = (lane >> 4) << 3;    // 0 or 8: K-half per ISA f16 layout
  const int rowBase = blockIdx.x * 128;

  floatx8 acc[8];
  const floatx8 zv = {0.f,0.f,0.f,0.f,0.f,0.f,0.f,0.f};

  auto stageW = [&](const _Float16* Wt, int K, int k0) {
    for (int i = tid; i < 512; i += 256) {
      int r = i >> 2, c = i & 3;
      copy16ToLds(Wt + (size_t)r * K + k0 + c * 8, &sW[r * APAD + c * 8]);
    }
  };

  // Load all 8 B fragments into distinct registers, then issue the 8 WMMAs
  // back-to-back behind a single dscnt drain (XDL pipe stays busy).
  auto wmmaPass = [&](half16 af) {
    half16 bf[8];
    #pragma unroll
    for (int t = 0; t < 8; ++t)
      bf[t] = ldsFrag(sW + (t * 16 + lr) * APAD + kh);
    #pragma unroll
    for (int t = 0; t < 8; ++t)
      acc[t] = __builtin_amdgcn_wmma_f32_16x16x32_f16(
          false, af, false, bf[t], (short)0, acc[t], false, false);
  };

  // ---------------- Layer 1: A gathered from global segments ---------------
  #pragma unroll
  for (int t = 0; t < 8; ++t) acc[t] = zv;

  for (int k0 = 0; k0 < a.KIN; k0 += 32) {
    const int s = k0 >> 7;
    const _Float16* base = (s == 0) ? a.seg0 : ((s == 1) ? a.seg1 : a.seg2);
    const int* ix        = (s == 0) ? (const int*)nullptr
                                    : ((s == 1) ? a.idx1 : a.idx2);
    const int kk = k0 & 127;
    for (int i = tid; i < 512; i += 256) {
      int r = i >> 2, c = i & 3;
      int e = rowBase + r;
      if (e < a.M) {
        size_t row = ix ? (size_t)ix[e] : (size_t)e;
        copy16ToLds(base + row * HID + kk + c * 8, &sA[r * APAD + c * 8]);
      } else {
        half8 z = {0,0,0,0,0,0,0,0};
        *(half8*)&sA[r * APAD + c * 8] = z;
      }
    }
    stageW(a.W1t, a.KIN, k0);
    stageWait();
    __syncthreads();

    wmmaPass(ldsFrag(sA + (wave * 16 + lr) * APAD + kh));
    __syncthreads();
  }

  auto reluToH = [&](const float* b) {
    if (tid < 128) sB[tid] = b[tid];
    __syncthreads();
    #pragma unroll
    for (int t = 0; t < 8; ++t) {
      int col = t * 16 + lr;
      #pragma unroll
      for (int r = 0; r < 8; ++r) {
        int row = wave * 16 + r + ((lane >> 4) << 3);
        float v = acc[t][r] + sB[col];
        sH[row * HPAD + col] = (_Float16)(v > 0.f ? v : 0.f);
      }
    }
    __syncthreads();
  };
  reluToH(a.b1);

  // ---------------- Layers 2 & 3: A from LDS hidden buffer -----------------
  auto gemmH = [&](const _Float16* Wt) {
    #pragma unroll
    for (int t = 0; t < 8; ++t) acc[t] = zv;
    for (int k0 = 0; k0 < HID; k0 += 32) {
      stageW(Wt, HID, k0);
      stageWait();
      __syncthreads();
      wmmaPass(ldsFrag(sH + (wave * 16 + lr) * HPAD + k0 + kh));
      __syncthreads();
    }
  };

  gemmH(a.W2t);
  reluToH(a.b2);
  gemmH(a.W3t);

  // ---------------- Epilogue: bias (+LN) (+residual) -> global -------------
  if (tid < 128) {
    sB[tid] = a.b3[tid];
    if (a.g) { sG[tid] = a.g[tid]; sBt[tid] = a.beta[tid]; }
  }
  __syncthreads();

  #pragma unroll
  for (int r = 0; r < 8; ++r) {
    int row = rowBase + wave * 16 + r + ((lane >> 4) << 3);
    float val[8];
    float s1 = 0.f, s2 = 0.f;
    #pragma unroll
    for (int t = 0; t < 8; ++t) {
      float v = acc[t][r] + sB[t * 16 + lr];
      val[t] = v; s1 += v; s2 += v * v;
    }
    if (a.g) {  // row-wise LayerNorm: reduce across the 16-lane half
      for (int m = 1; m < 16; m <<= 1) {
        s1 += __shfl_xor(s1, m, 32);
        s2 += __shfl_xor(s2, m, 32);
      }
      float mu   = s1 * (1.f / 128.f);
      float var  = s2 * (1.f / 128.f) - mu * mu;
      float rstd = rsqrtf(var + LNEPS);
      #pragma unroll
      for (int t = 0; t < 8; ++t) {
        int col = t * 16 + lr;
        val[t] = (val[t] - mu) * rstd * sG[col] + sBt[col];
      }
    }
    if (row < a.M) {
      #pragma unroll
      for (int t = 0; t < 8; ++t) {
        int col = t * 16 + lr;
        float v = val[t];
        if (a.resid) v += a.resid[(size_t)row * HID + col];
        a.outF[(size_t)row * HID + col] = v;
        if (a.outH) a.outH[(size_t)row * HID + col] = (_Float16)v;
      }
    }
  }
}

// --------------------------- support kernels --------------------------------

// fp32 [count][K][N] -> f16 transposed+padded [count][NP][KP]
__global__ void prep_wt(const float* src, _Float16* dst,
                        int K, int N, int KP, int NP, int count) {
  long total = (long)count * NP * KP;
  for (long idx = blockIdx.x * 256L + threadIdx.x; idx < total;
       idx += (long)gridDim.x * 256L) {
    long s = idx / ((long)NP * KP);
    long r = idx - s * (long)NP * KP;
    int  n = (int)(r / KP);
    int  k = (int)(r - (long)n * KP);
    float v = (n < N && k < K) ? src[(s * K + k) * (long)N + n] : 0.f;
    dst[idx] = (_Float16)v;
  }
}

__global__ void pad_bias(const float* src, float* dst, int n, int np) {
  int i = blockIdx.x * 256 + threadIdx.x;
  if (i < np) dst[i] = (i < n) ? src[i] : 0.f;
}

__global__ void node_features(const float* wp, const float* pwp,
                              const int* ntype, const float* mean,
                              const float* stdv, _Float16* out, int N) {
  long i = blockIdx.x * 256L + threadIdx.x;
  if (i >= (long)N * 32) return;
  int n = (int)(i >> 5), c = (int)(i & 31);
  float v = 0.f;
  if (c < 12) {
    float raw = (c < 3) ? (wp[n * 3 + c] - pwp[n * 3 + c])
                        : ((ntype[n] == (c - 3)) ? 1.f : 0.f);
    v = (raw - mean[c]) / stdv[c];
  }
  out[(size_t)n * HID + c] = (_Float16)v;
}

__global__ void edge_features(const float* wp, const float* mp,
                              const int* srcI, const int* dstI,
                              const float* mean, const float* stdv,
                              _Float16* out, int E) {
  int e = blockIdx.x * 256 + threadIdx.x;
  if (e >= E) return;
  int s = srcI[e], d = dstI[e];
  float rw[3], rm[3], nw = 0.f, nm = 0.f;
  for (int c = 0; c < 3; ++c) {
    rw[c] = wp[s * 3 + c] - wp[d * 3 + c];  nw += rw[c] * rw[c];
    rm[c] = mp[s * 3 + c] - mp[d * 3 + c];  nm += rm[c] * rm[c];
  }
  float f[8] = { rw[0], rw[1], rw[2], sqrtf(nw),
                 rm[0], rm[1], rm[2], sqrtf(nm) };
  _Float16* row = out + (size_t)e * HID;
  for (int c = 0; c < 8; ++c)  row[c] = (_Float16)((f[c] - mean[c]) / stdv[c]);
  for (int c = 8; c < 32; ++c) row[c] = (_Float16)0.f;
}

__global__ void zero_f32(float* p, long n) {
  long i = blockIdx.x * 256L + threadIdx.x;
  if (i < n) p[i] = 0.f;
}

__global__ void scatter_add(const float* e_new, const int* dstI,
                            float* agg, int E) {
  long i = blockIdx.x * 256L + threadIdx.x;
  if (i >= (long)E * HID) return;
  int e = (int)(i >> 7), c = (int)(i & 127);
  int d = dstI[e];
  __hip_atomic_fetch_add(&agg[(size_t)d * HID + c], e_new[i],
                         __ATOMIC_RELAXED, __HIP_MEMORY_SCOPE_AGENT);
}

__global__ void f32_to_f16(const float* s, _Float16* d, long n) {
  long i = blockIdx.x * 256L + threadIdx.x;
  if (i < n) d[i] = (_Float16)s[i];
}

__global__ void integrate(const float* dec_out, const float* wp,
                          const float* pwp, const int* ntype,
                          const float* ostd, const float* omean,
                          float* out, int N) {
  int n = blockIdx.x * 256 + threadIdx.x;
  if (n >= N) return;
  bool normal = (ntype[n] == 0);
  for (int c = 0; c < 3; ++c) {
    float acc  = dec_out[(size_t)n * HID + c] * ostd[c] + omean[c];
    float pred = 2.f * wp[n * 3 + c] + acc - pwp[n * 3 + c];
    out[n * 3 + c] = normal ? pred : wp[n * 3 + c];
  }
}

// ------------------------------- launcher -----------------------------------

extern "C" void kernel_launch(void* const* d_in, const int* in_sizes, int n_in,
                              void* d_out, int out_size, void* d_ws, size_t ws_size,
                              hipStream_t stream) {
  (void)n_in; (void)out_size; (void)ws_size;
  const int N = in_sizes[0] / 3;
  const int E = in_sizes[4] / 2;

  // Input indices (setup_inputs dict order, params flattened in insertion
  // order: mean/std scalars, enc_node{W1,b1,W2,b2,W3,b3,g,beta}, enc_edge{..},
  // blk_edge{..}, blk_node{..}, dec{W1,b1,W2,b2,W3,b3}).
  const float* wp    = (const float*)d_in[0];
  const float* pwp   = (const float*)d_in[1];
  const float* mp    = (const float*)d_in[2];
  const int*   ntype = (const int*)d_in[3];
  const int*   srcI  = (const int*)d_in[4];
  const int*   dstI  = srcI + E;
  const float* nodeMean = (const float*)d_in[5];
  const float* nodeStd  = (const float*)d_in[6];
  const float* edgeMean = (const float*)d_in[7];
  const float* edgeStd  = (const float*)d_in[8];
  const float* outMean  = (const float*)d_in[9];
  const float* outStd   = (const float*)d_in[10];

  // workspace carve-out
  char* ws = (char*)d_ws;
  size_t off = 0;
  auto alloc = [&](size_t bytes) {
    void* p = ws + off;
    off = (off + bytes + 255) & ~(size_t)255;
    return p;
  };
  _Float16* encNodeW1t = (_Float16*)alloc((size_t)128 * 32 * 2);
  _Float16* encNodeW2t = (_Float16*)alloc((size_t)128 * 128 * 2);
  _Float16* encNodeW3t = (_Float16*)alloc((size_t)128 * 128 * 2);
  _Float16* encEdgeW1t = (_Float16*)alloc((size_t)128 * 32 * 2);
  _Float16* encEdgeW2t = (_Float16*)alloc((size_t)128 * 128 * 2);
  _Float16* encEdgeW3t = (_Float16*)alloc((size_t)128 * 128 * 2);
  _Float16* beW1t = (_Float16*)alloc((size_t)15 * 128 * 384 * 2);
  _Float16* beW2t = (_Float16*)alloc((size_t)15 * 128 * 128 * 2);
  _Float16* beW3t = (_Float16*)alloc((size_t)15 * 128 * 128 * 2);
  _Float16* bnW1t = (_Float16*)alloc((size_t)15 * 128 * 256 * 2);
  _Float16* bnW2t = (_Float16*)alloc((size_t)15 * 128 * 128 * 2);
  _Float16* bnW3t = (_Float16*)alloc((size_t)15 * 128 * 128 * 2);
  _Float16* decW1t = (_Float16*)alloc((size_t)128 * 128 * 2);
  _Float16* decW2t = (_Float16*)alloc((size_t)128 * 128 * 2);
  _Float16* decW3t = (_Float16*)alloc((size_t)128 * 128 * 2);
  float*    decB3p = (float*)alloc(128 * 4);
  _Float16* nodeFeatH = (_Float16*)alloc((size_t)N * HID * 2);
  _Float16* edgeFeatH = (_Float16*)alloc((size_t)E * HID * 2);
  float*    nodeLat   = (float*)alloc((size_t)N * HID * 4);
  _Float16* nodeLatH  = (_Float16*)alloc((size_t)N * HID * 2);
  float*    edgeLat   = (float*)alloc((size_t)E * HID * 4);
  _Float16* edgeLatH  = (_Float16*)alloc((size_t)E * HID * 2);
  float*    agg       = (float*)alloc((size_t)N * HID * 4);
  _Float16* aggH      = (_Float16*)alloc((size_t)N * HID * 2);
  float*    decOut    = (float*)alloc((size_t)N * HID * 4);

  auto prep = [&](int inIdx, _Float16* dst, int K, int Nn, int KP, int NP, int cnt) {
    long total = (long)cnt * NP * KP;
    int blocks = (int)((total + 255) / 256);
    prep_wt<<<blocks, 256, 0, stream>>>((const float*)d_in[inIdx], dst,
                                        K, Nn, KP, NP, cnt);
  };
  prep(11, encNodeW1t, 12, 128, 32, 128, 1);
  prep(13, encNodeW2t, 128, 128, 128, 128, 1);
  prep(15, encNodeW3t, 128, 128, 128, 128, 1);
  prep(19, encEdgeW1t, 8, 128, 32, 128, 1);
  prep(21, encEdgeW2t, 128, 128, 128, 128, 1);
  prep(23, encEdgeW3t, 128, 128, 128, 128, 1);
  prep(27, beW1t, 384, 128, 384, 128, 15);
  prep(29, beW2t, 128, 128, 128, 128, 15);
  prep(31, beW3t, 128, 128, 128, 128, 15);
  prep(35, bnW1t, 256, 128, 256, 128, 15);
  prep(37, bnW2t, 128, 128, 128, 128, 15);
  prep(39, bnW3t, 128, 128, 128, 128, 15);
  prep(43, decW1t, 128, 128, 128, 128, 1);
  prep(45, decW2t, 128, 128, 128, 128, 1);
  prep(47, decW3t, 128, 3, 128, 128, 1);
  pad_bias<<<1, 256, 0, stream>>>((const float*)d_in[48], decB3p, 3, 128);

  node_features<<<(int)(((long)N * 32 + 255) / 256), 256, 0, stream>>>(
      wp, pwp, ntype, nodeMean, nodeStd, nodeFeatH, N);
  edge_features<<<(E + 255) / 256, 256, 0, stream>>>(
      wp, mp, srcI, dstI, edgeMean, edgeStd, edgeFeatH, E);

  auto runMlp = [&](MlpArgs a) {
    int blocks = (a.M + 127) / 128;
    mlp3_wmma<<<blocks, 256, 0, stream>>>(a);
  };

  // encoders
  {
    MlpArgs a = {};
    a.seg0 = nodeFeatH; a.KIN = 32; a.M = N;
    a.W1t = encNodeW1t; a.W2t = encNodeW2t; a.W3t = encNodeW3t;
    a.b1 = (const float*)d_in[12]; a.b2 = (const float*)d_in[14];
    a.b3 = (const float*)d_in[16];
    a.g = (const float*)d_in[17]; a.beta = (const float*)d_in[18];
    a.outF = nodeLat; a.outH = nodeLatH;
    runMlp(a);
  }
  {
    MlpArgs a = {};
    a.seg0 = edgeFeatH; a.KIN = 32; a.M = E;
    a.W1t = encEdgeW1t; a.W2t = encEdgeW2t; a.W3t = encEdgeW3t;
    a.b1 = (const float*)d_in[20]; a.b2 = (const float*)d_in[22];
    a.b3 = (const float*)d_in[24];
    a.g = (const float*)d_in[25]; a.beta = (const float*)d_in[26];
    a.outF = edgeLat; a.outH = edgeLatH;
    runMlp(a);
  }

  // 15 graph-net blocks
  for (int b = 0; b < 15; ++b) {
    {   // edge MLP: [el | nl[src] | nl[dst]] gathered in-kernel
      MlpArgs a = {};
      a.seg0 = edgeLatH; a.seg1 = nodeLatH; a.seg2 = nodeLatH;
      a.idx1 = srcI; a.idx2 = dstI;
      a.KIN = 384; a.M = E;
      a.W1t = beW1t + (size_t)b * 128 * 384;
      a.W2t = beW2t + (size_t)b * 128 * 128;
      a.W3t = beW3t + (size_t)b * 128 * 128;
      a.b1 = (const float*)d_in[28] + b * 128;
      a.b2 = (const float*)d_in[30] + b * 128;
      a.b3 = (const float*)d_in[32] + b * 128;
      a.g  = (const float*)d_in[33] + b * 128;
      a.beta = (const float*)d_in[34] + b * 128;
      a.resid = edgeLat; a.outF = edgeLat; a.outH = edgeLatH;
      runMlp(a);
    }
    zero_f32<<<(int)(((long)N * HID + 255) / 256), 256, 0, stream>>>(
        agg, (long)N * HID);
    scatter_add<<<(int)(((long)E * HID + 255) / 256), 256, 0, stream>>>(
        edgeLat, dstI, agg, E);
    f32_to_f16<<<(int)(((long)N * HID + 255) / 256), 256, 0, stream>>>(
        agg, aggH, (long)N * HID);
    {   // node MLP: [nl | agg]
      MlpArgs a = {};
      a.seg0 = nodeLatH; a.seg1 = aggH;
      a.KIN = 256; a.M = N;
      a.W1t = bnW1t + (size_t)b * 128 * 256;
      a.W2t = bnW2t + (size_t)b * 128 * 128;
      a.W3t = bnW3t + (size_t)b * 128 * 128;
      a.b1 = (const float*)d_in[36] + b * 128;
      a.b2 = (const float*)d_in[38] + b * 128;
      a.b3 = (const float*)d_in[40] + b * 128;
      a.g  = (const float*)d_in[41] + b * 128;
      a.beta = (const float*)d_in[42] + b * 128;
      a.resid = nodeLat; a.outF = nodeLat; a.outH = nodeLatH;
      runMlp(a);
    }
  }

  // decoder (no LN, W3 zero-padded to 128 cols; cols 0..2 are the output)
  {
    MlpArgs a = {};
    a.seg0 = nodeLatH; a.KIN = 128; a.M = N;
    a.W1t = decW1t; a.W2t = decW2t; a.W3t = decW3t;
    a.b1 = (const float*)d_in[44]; a.b2 = (const float*)d_in[46];
    a.b3 = decB3p;
    a.outF = decOut;
    runMlp(a);
  }

  integrate<<<(N + 255) / 256, 256, 0, stream>>>(
      decOut, wp, pwp, ntype, outStd, outMean, (float*)d_out, N);
}
